// DAM_45200235823221
// MI455X (gfx1250) — matrix operand order repaired and checked by reading, not compiled, not run on hardware
//
#include <hip/hip_runtime.h>
#include <hip/hip_bf16.h>
#include <math.h>

// ---------------- types ----------------
typedef __attribute__((ext_vector_type(16))) __bf16 v16bf;
typedef __attribute__((ext_vector_type(8)))  float  v8f;

union Frag { v16bf v; uint4 q[2]; };

#define DIMD 128   // embed dim
#define DIM2 256   // 2*d
#define LMAX 100   // max items per bundle

// ---------------- fp32 -> bf16 convert ----------------
__global__ __launch_bounds__(256) void cvt_bf16_kernel(
    const float* __restrict__ in, __hip_bfloat16* __restrict__ out, int n) {
  int i = blockIdx.x * 256 + threadIdx.x;
  if (i < n) out[i] = __float2bfloat16(in[i]);
}

// ---------------- fused gather + attention + concat ----------------
// one block (8 waves) per bundle; writes h = [h_u | h_b + h_x] as bf16 [B,256]
__global__ __launch_bounds__(256) void attn_kernel(
    const int* __restrict__ x_u, const int* __restrict__ x_b,
    const int* __restrict__ items, const unsigned char* __restrict__ mask, // jax bool = 1 byte
    const float* __restrict__ emb_u, const float* __restrict__ emb_i,
    const float* __restrict__ emb_b, const float* __restrict__ A,
    __hip_bfloat16* __restrict__ hb) {
  __shared__ float hu_s[DIMD];
  __shared__ float sc_s[128];   // scores, padded to 128 with -inf
  __shared__ int   idx_s[128];

  const int b    = blockIdx.x;
  const int t    = threadIdx.x;
  const int wave = t >> 5;
  const int lane = t & 31;

  const int u = x_u[b];
  if (t < DIMD) hu_s[t] = emb_u[(size_t)u * DIMD + t];
  if (t < 128)  sc_s[t] = -INFINITY;
  if (t < LMAX) idx_s[t] = items[(size_t)b * LMAX + t];
  __syncthreads();

  // scores: one wave per item, lane owns 4 contiguous floats (16B vector loads)
  for (int l = wave; l < LMAX; l += 8) {
    const float4* arow = (const float4*)(A + (size_t)idx_s[l] * DIMD);
    const float4  av   = arow[lane];
    const float4  hv   = *(const float4*)&hu_s[lane * 4];
    float p = av.x * hv.x + av.y * hv.y + av.z * hv.z + av.w * hv.w;
#pragma unroll
    for (int off = 16; off > 0; off >>= 1) p += __shfl_xor(p, off, 32);
    if (lane == 0)
      sc_s[l] = mask[(size_t)b * LMAX + l] ? p : -INFINITY;
  }
  __syncthreads();

  // masked softmax over 100 scores: single wave, 4 elements per lane
  if (wave == 0) {
    float v[4];
    float mx = -INFINITY;
#pragma unroll
    for (int j = 0; j < 4; ++j) { v[j] = sc_s[lane + 32 * j]; mx = fmaxf(mx, v[j]); }
#pragma unroll
    for (int off = 16; off > 0; off >>= 1) mx = fmaxf(mx, __shfl_xor(mx, off, 32));
    float s = 0.f;
#pragma unroll
    for (int j = 0; j < 4; ++j) { v[j] = __expf(v[j] - mx); s += v[j]; }  // exp(-inf)=0
#pragma unroll
    for (int off = 16; off > 0; off >>= 1) s += __shfl_xor(s, off, 32);
    const float inv = 1.f / s;
#pragma unroll
    for (int j = 0; j < 4; ++j) sc_s[lane + 32 * j] = v[j] * inv;
  }
  __syncthreads();

  // h_x[d] = sum_l w[l] * emb_i[item_l][d]; then concat + bf16 store
  if (t < DIMD) {
    float acc = 0.f;
    for (int l = 0; l < LMAX; ++l)
      acc += sc_s[l] * emb_i[(size_t)idx_s[l] * DIMD + t];
    const float hbd = emb_b[(size_t)x_b[b] * DIMD + t];
    hb[(size_t)b * DIM2 + t]        = __float2bfloat16(hu_s[t]);
    hb[(size_t)b * DIM2 + DIMD + t] = __float2bfloat16(hbd + acc);
  }
}

// ---------------- WMMA fragment loads (contiguous k order, same in A and B) ----------------
// A (16x32 bf16, M x K): lane<16 -> row M=lane, K = k0..k0+15
//                        lane>=16 -> row M=lane-16, K = k0+16..k0+31
__device__ inline v16bf load_afrag(const __hip_bfloat16* X, int ldk, int row0, int k0, int lane) {
  const int r  = row0 + (lane & 15);
  const int kb = k0 + ((lane & 16) ? 16 : 0);
  const char* base = (const char*)(X + (size_t)r * ldk + kb);
  Frag f;
  f.q[0] = *(const uint4*)(base);        // K = kb .. kb+7
  f.q[1] = *(const uint4*)(base + 16);   // K = kb+8 .. kb+15
  return f.v;
}
// B (32x16 bf16, K x N) with B[k][n] = W[n][k]: lane<16 -> col N=lane, K=k0..k0+15
//                                               lane>=16 -> col N=lane-16, K=k0+16..k0+31
// => one contiguous 32-byte row segment of W per lane.
__device__ inline v16bf load_bfrag(const __hip_bfloat16* W, int ldk, int col0, int k0, int lane) {
  const int c  = col0 + (lane & 15);
  const int kb = k0 + ((lane & 16) ? 16 : 0);
  const uint4* p = (const uint4*)(W + (size_t)c * ldk + kb);
  Frag f;
  f.q[0] = p[0];
  f.q[1] = p[1];
  return f.v;
}

// ---------------- bf16 WMMA GEMM: Y = act(X @ W^T + bias) ----------------
// X [M x K] bf16, W [N x K] bf16, K = N = 256. Block tile 128x64, 8 waves of 32x32.
__global__ __launch_bounds__(256) void gemm_kernel(
    const __hip_bfloat16* __restrict__ Xb, const __hip_bfloat16* __restrict__ Wb,
    const float* __restrict__ bias,
    float* __restrict__ Yf, __hip_bfloat16* __restrict__ Yb,
    int M, int N, int K, int apply_act) {
  const int lane = threadIdx.x & 31;
  const int wave = threadIdx.x >> 5;
  const int wm = wave & 3;          // 4 waves along M
  const int wn = wave >> 2;         // 2 waves along N
  const int m0 = blockIdx.x * 128 + wm * 32;   // wave's 32-row strip
  const int n0 = blockIdx.y * 64  + wn * 32;   // wave's 32-col strip

  v8f acc[2][2] = {};   // 2x2 grid of 16x16 f32 accumulator tiles

  for (int k0 = 0; k0 < K; k0 += 32) {
    v16bf a0 = load_afrag(Xb, K, m0,      k0, lane);
    v16bf a1 = load_afrag(Xb, K, m0 + 16, k0, lane);
    v16bf b0 = load_bfrag(Wb, K, n0,      k0, lane);
    v16bf b1 = load_bfrag(Wb, K, n0 + 16, k0, lane);
    acc[0][0] = __builtin_amdgcn_wmma_f32_16x16x32_bf16(false, a0, false, b0, (short)0, acc[0][0], false, false);
    acc[0][1] = __builtin_amdgcn_wmma_f32_16x16x32_bf16(false, a0, false, b1, (short)0, acc[0][1], false, false);
    acc[1][0] = __builtin_amdgcn_wmma_f32_16x16x32_bf16(false, a1, false, b0, (short)0, acc[1][0], false, false);
    acc[1][1] = __builtin_amdgcn_wmma_f32_16x16x32_bf16(false, a1, false, b1, (short)0, acc[1][1], false, false);
  }

  // C/D layout: lane<16 -> (M=r, N=lane), lane>=16 -> (M=r+8, N=lane-16), r = VGPR 0..7
  const int nc   = (lane & 15);
  const int mofs = (lane & 16) ? 8 : 0;
#pragma unroll
  for (int mi = 0; mi < 2; ++mi) {
#pragma unroll
    for (int ni = 0; ni < 2; ++ni) {
      const int n = n0 + ni * 16 + nc;
      const float bv = bias[n];
#pragma unroll
      for (int r = 0; r < 8; ++r) {
        const int m = m0 + mi * 16 + mofs + r;
        float v = acc[mi][ni][r] + bv;
        if (apply_act) v = (v > 0.f) ? v : 0.01f * v;   // jax leaky_relu slope
        if (Yf) Yf[(size_t)m * N + n] = v;
        if (Yb) Yb[(size_t)m * N + n] = __float2bfloat16(v);
      }
    }
  }
}

// ---------------- final GEMV: out[b] = h2[b,:] . out_w + out_b ----------------
__global__ __launch_bounds__(256) void outvec_kernel(
    const float* __restrict__ H, const float* __restrict__ ow,
    const float* __restrict__ ob, float* __restrict__ out, int M) {
  const int row  = blockIdx.x * 8 + (threadIdx.x >> 5);
  const int lane = threadIdx.x & 31;
  if (row >= M) return;
  float p = 0.f;
#pragma unroll
  for (int j = lane; j < DIM2; j += 32) p += H[(size_t)row * DIM2 + j] * ow[j];
#pragma unroll
  for (int off = 16; off > 0; off >>= 1) p += __shfl_xor(p, off, 32);
  if (lane == 0) out[row] = p + ob[0];
}

// ---------------- launcher ----------------
extern "C" void kernel_launch(void* const* d_in, const int* in_sizes, int n_in,
                              void* d_out, int out_size, void* d_ws, size_t ws_size,
                              hipStream_t stream) {
  const int* x_u   = (const int*)d_in[0];
  const int* x_b   = (const int*)d_in[1];
  const int* items = (const int*)d_in[2];
  const unsigned char* mask = (const unsigned char*)d_in[3];  // jax bool -> 1 byte
  const float* emb_u = (const float*)d_in[4];
  const float* emb_i = (const float*)d_in[5];
  const float* emb_b = (const float*)d_in[6];
  const float* A     = (const float*)d_in[7];
  const float* fc1_w = (const float*)d_in[8];
  const float* fc1_b = (const float*)d_in[9];
  const float* fc2_w = (const float*)d_in[10];
  const float* fc2_b = (const float*)d_in[11];
  const float* out_w = (const float*)d_in[12];
  const float* out_b = (const float*)d_in[13];
  float* out = (float*)d_out;

  const int B = in_sizes[0];          // 8192 (multiple of 128)

  // workspace carve-up
  char* ws = (char*)d_ws;
  __hip_bfloat16* hb  = (__hip_bfloat16*)(ws);                               // [B,256] bf16
  __hip_bfloat16* h1b = (__hip_bfloat16*)(ws + (size_t)B * DIM2 * 2);        // [B,256] bf16
  float*          h2f = (float*)        (ws + (size_t)B * DIM2 * 4);         // [B,256] f32
  __hip_bfloat16* w1b = (__hip_bfloat16*)(ws + (size_t)B * DIM2 * 8);        // [256,256] bf16
  __hip_bfloat16* w2b = w1b + (size_t)DIM2 * DIM2;                           // [256,256] bf16

  // 1) weights fp32 -> bf16
  const int WN = DIM2 * DIM2;
  cvt_bf16_kernel<<<(WN + 255) / 256, 256, 0, stream>>>(fc1_w, w1b, WN);
  cvt_bf16_kernel<<<(WN + 255) / 256, 256, 0, stream>>>(fc2_w, w2b, WN);

  // 2) fused gathers + attention + concat -> hb (bf16)
  attn_kernel<<<B, 256, 0, stream>>>(x_u, x_b, items, mask, emb_u, emb_i, emb_b, A, hb);

  // 3) fc1: h1 = leaky_relu(hb @ w1^T + b1)   (bf16 out for next GEMM)
  dim3 g1(B / 128, DIM2 / 64);
  gemm_kernel<<<g1, 256, 0, stream>>>(hb, w1b, fc1_b, nullptr, h1b, B, DIM2, DIM2, 1);

  // 4) fc2: h2 = leaky_relu(h1 @ w2^T + b2)   (f32 out for GEMV)
  gemm_kernel<<<g1, 256, 0, stream>>>(h1b, w2b, fc2_b, h2f, nullptr, B, DIM2, DIM2, 1);

  // 5) out = h2 @ out_w^T + out_b
  outvec_kernel<<<(B + 7) / 8, 256, 0, stream>>>(h2f, out_w, out_b, out, B);
}
